// SelfAttentionClassifier_40578851013046
// MI455X (gfx1250) — matrix operand order repaired and checked by reading, not compiled, not run on hardware
//
#include <hip/hip_runtime.h>

typedef __attribute__((ext_vector_type(16))) _Float16 v16h;
typedef __attribute__((ext_vector_type(8)))  float    v8f;
typedef int a4i __attribute__((__vector_size__(4 * sizeof(int))));

#define D_DIM 1024
#define S_LEN 2048
#define B_SZ  8
#define NCLS  2

#if defined(__has_builtin)
#  if __has_builtin(__builtin_amdgcn_global_load_async_to_lds_b128)
#    define USE_ASYNC_LDS 1
#  endif
#endif
#ifndef USE_ASYNC_LDS
#  define USE_ASYNC_LDS 0
#endif

#if defined(__has_builtin)
#  if __has_builtin(__builtin_amdgcn_s_wait_asynccnt)
#    define WAIT_ASYNC() __builtin_amdgcn_s_wait_asynccnt(0)
#  endif
#endif
#ifndef WAIT_ASYNC
#  define WAIT_ASYNC() asm volatile("s_wait_asynccnt 0x0" ::: "memory")
#endif

// ---------------- fp32 -> f16 conversion ----------------
__global__ __launch_bounds__(256) void cvt_f32_f16(const float* __restrict__ src,
                                                   _Float16* __restrict__ dst, int n4) {
  int i = blockIdx.x * blockDim.x + threadIdx.x;
  if (i < n4) {
    float4 f = ((const float4*)src)[i];
    union { _Float16 h[4]; uint2 u; } t;
    t.h[0] = (_Float16)f.x; t.h[1] = (_Float16)f.y;
    t.h[2] = (_Float16)f.z; t.h[3] = (_Float16)f.w;
    *(uint2*)(dst + 4 * (size_t)i) = t.u;
  }
}

// ---------------- f16 WMMA GEMM: C[M,N](f16) = A[M,K] @ B[K,N] + bias ----------------
#define BM 128
#define BN 128
#define BK 32
#define LDSS 40   // padded LDS row stride in halfs (80B, 16B aligned)

// one K-step of WMMAs on a committed LDS stage (LDS bases constant per call site)
__device__ __forceinline__ void mma_step(const _Float16* __restrict__ sAst,
                                         const _Float16* __restrict__ sBst,
                                         int wr, int wc, int lrow, int khalf, int k0,
                                         v8f acc[2][4]) {
  union Frag { v16h v; uint4 u[2]; };
  Frag fa[2];
  #pragma unroll
  for (int i = 0; i < 2; ++i) {
    int row = wr * 32 + i * 16 + lrow;
    // A operand: lane-half khalf -> K = k0..k0+7 and k0+16..k0+23
    fa[i].u[0] = *(const uint4*)(sAst + row * LDSS + k0);
    fa[i].u[1] = *(const uint4*)(sAst + row * LDSS + k0 + 16);
  }
  #pragma unroll
  for (int j = 0; j < 4; ++j) {
    int col = wc * 64 + j * 16 + lrow;
    // B operand: lane holds column `col`, K = k0..k0+7 and k0+16..k0+23
    Frag fb;
    fb.u[0] = *(const uint4*)(sBst + col * LDSS + k0);
    fb.u[1] = *(const uint4*)(sBst + col * LDSS + k0 + 16);
    #pragma unroll
    for (int i = 0; i < 2; ++i) {
      acc[i][j] = __builtin_amdgcn_wmma_f32_16x16x32_f16(
          false, fa[i].v, false, fb.v, (short)0, acc[i][j], false, false);
    }
  }
}

__global__ __launch_bounds__(256) void gemm_f16_wmma(
    const _Float16* __restrict__ A, const _Float16* __restrict__ Bw,
    const float* __restrict__ bias, _Float16* __restrict__ C,
    int M, int N, int K)
{
  __shared__ _Float16 sA[2][BM * LDSS];
  __shared__ _Float16 sB[2][BN * LDSS];   // transposed: sB[n][k]

  const int tid   = threadIdx.x;
  const int wid   = tid >> 5;          // wave32
  const int lane  = tid & 31;
  const int mbase = blockIdx.y * BM;
  const int nbase = blockIdx.x * BN;
  const int wr    = wid >> 1;          // 0..3 -> 32-row group
  const int wc    = wid & 1;           // 0..1 -> 64-col group
  const int lrow  = lane & 15;
  const int khalf = lane >> 4;
  const int k0    = khalf * 8;

  // thread-dependent parts of addresses, 32-bit, computed once
  const int aRow0 = (tid * 8) >> 5,        aK0 = (tid * 8) & 31;
  const int aRow1 = ((tid + 256) * 8) >> 5, aK1 = ((tid + 256) * 8) & 31;
  const int aOff0 = (mbase + aRow0) * K + aK0;     // element offset into A
  const int aOff1 = (mbase + aRow1) * K + aK1;
  const int aLds0 = aRow0 * LDSS + aK0;
  const int aLds1 = aRow1 * LDSS + aK1;
  const int nG = tid & 15;                 // B staging: 8-col group
  const int kP = tid >> 4;                 // B staging: K-row pair
  const int bOff = 2 * kP * N + nbase + nG * 8;    // element offset into Bw (+ kb*N uniform)
  const int bLds = nG * 8 * LDSS + 2 * kP;

  v8f acc[2][4];
  const v8f vzero = {0.f,0.f,0.f,0.f,0.f,0.f,0.f,0.f};
  #pragma unroll
  for (int i = 0; i < 2; ++i)
    #pragma unroll
    for (int j = 0; j < 4; ++j) acc[i][j] = vzero;

  uint4 pb0, pb1;
#if !USE_ASYNC_LDS
  uint4 pa0, pa1;
#endif

#define FETCH_B(kb) do { const _Float16* Bk = Bw + (size_t)(kb) * N;                  \
    pb0 = *(const uint4*)(Bk + bOff); pb1 = *(const uint4*)(Bk + bOff + N); } while(0)

#if USE_ASYNC_LDS
#define FETCH_A(kb, stg) do { const _Float16* Ak = A + (kb);                          \
    __builtin_amdgcn_global_load_async_to_lds_b128((a4i*)(Ak + aOff0),                \
        (a4i*)(&sA[stg][aLds0]), 0, 0);                                               \
    __builtin_amdgcn_global_load_async_to_lds_b128((a4i*)(Ak + aOff1),                \
        (a4i*)(&sA[stg][aLds1]), 0, 0); } while(0)
#define COMMIT_A(stg) WAIT_ASYNC()
#else
#define FETCH_A(kb, stg) do { const _Float16* Ak = A + (kb);                          \
    pa0 = *(const uint4*)(Ak + aOff0); pa1 = *(const uint4*)(Ak + aOff1); } while(0)
#define COMMIT_A(stg) do { *(uint4*)(&sA[stg][aLds0]) = pa0;                          \
    *(uint4*)(&sA[stg][aLds1]) = pa1; } while(0)
#endif

#define COMMIT_B(stg) do {                                                            \
    union { uint4 u; _Float16 h[8]; } b0_, b1_; b0_.u = pb0; b1_.u = pb1;             \
    _Pragma("unroll")                                                                 \
    for (int e = 0; e < 8; ++e) {                                                     \
      union { _Float16 h[2]; unsigned int u; } p_;                                    \
      p_.h[0] = b0_.h[e]; p_.h[1] = b1_.h[e];                                         \
      *(unsigned int*)(&sB[stg][bLds + e * LDSS]) = p_.u;                             \
    } } while(0)

  // ---- prologue: begin fetching tile kb=0 into stage 0
  FETCH_B(0);
  FETCH_A(0, 0);

  for (int kb = 0; kb < K; kb += 2 * BK) {
    // ================= stage 0: holds tile kb =================
    COMMIT_B(0);
    COMMIT_A(0);
    __syncthreads();
    FETCH_B(kb + BK);              // kb+BK < K always (K multiple of 2*BK)
    FETCH_A(kb + BK, 1);
    mma_step(sA[0], sB[0], wr, wc, lrow, khalf, k0, acc);

    // ================= stage 1: holds tile kb+BK ==============
    COMMIT_B(1);
    COMMIT_A(1);
    __syncthreads();
    const int kn = (kb + 2 * BK < K) ? (kb + 2 * BK) : (kb + BK);  // clamp on last
    FETCH_B(kn);
    FETCH_A(kn, 0);
    mma_step(sA[1], sB[1], wr, wc, lrow, khalf, k0, acc);
  }

#undef FETCH_B
#undef FETCH_A
#undef COMMIT_A
#undef COMMIT_B

  // epilogue: +bias, store f16. C/D layout: VGPR r -> M=r (lanes 0-15), M=8+r (lanes 16-31)
  #pragma unroll
  for (int i = 0; i < 2; ++i) {
    #pragma unroll
    for (int j = 0; j < 4; ++j) {
      int n = nbase + wc * 64 + j * 16 + lrow;
      float bvv = bias[n];
      #pragma unroll
      for (int r = 0; r < 8; ++r) {
        int m = mbase + wr * 32 + i * 16 + r + 8 * khalf;
        C[(size_t)m * N + n] = (_Float16)(acc[i][j][r] + bvv);
      }
    }
  }
}

// ---------------- 8-row projection: out[b,n] = dot(rows[b,:], W[:,n]) + bias (opt relu) ----------------
__global__ __launch_bounds__(256) void proj_row(
    const float* __restrict__ rows, long rowStride,
    const float* __restrict__ W, const float* __restrict__ bias,
    float* __restrict__ out, int N, int K, int doRelu)
{
  __shared__ float sx[D_DIM];
  int b = blockIdx.y;
  int n = blockIdx.x * 256 + threadIdx.x;
  const float* xr = rows + (long)b * rowStride;
  for (int k = threadIdx.x; k < K; k += 256) sx[k] = xr[k];
  __syncthreads();
  float acc = 0.f;
  for (int k = 0; k < K; ++k) acc += sx[k] * W[(size_t)k * N + n];
  acc += bias[n];
  if (doRelu) acc = fmaxf(acc, 0.f);
  out[(size_t)b * N + n] = acc;
}

// ---------------- scores[b,t] = dot(q0[b,:], K[b,t,:]) / 32 ----------------
__global__ __launch_bounds__(256) void attn_scores(
    const float* __restrict__ q0, const _Float16* __restrict__ kh,
    float* __restrict__ scores)
{
  int wid  = threadIdx.x >> 5;
  int lane = threadIdx.x & 31;
  int key  = blockIdx.x * 8 + wid;         // flat (b,t)
  int b    = key >> 11;                    // /2048
  const float*    q  = q0 + b * D_DIM;
  const _Float16* kv = kh + (size_t)key * D_DIM;
  float acc = 0.f;
  #pragma unroll 4
  for (int i = 0; i < 32; ++i) {
    int idx = i * 32 + lane;
    acc += q[idx] * (float)kv[idx];
  }
  for (int off = 16; off > 0; off >>= 1) acc += __shfl_down(acc, off);
  if (lane == 0) scores[key] = acc * 0.03125f;  // 1/sqrt(1024)
}

// ---------------- softmax over t (2048) per batch ----------------
__global__ __launch_bounds__(256) void attn_softmax(const float* __restrict__ scores,
                                                    float* __restrict__ probs)
{
  __shared__ float red[256];
  int b = blockIdx.x;
  const float* s = scores + (size_t)b * S_LEN;
  float v[8];
  float mx = -1e30f;
  #pragma unroll
  for (int i = 0; i < 8; ++i) { v[i] = s[threadIdx.x + i * 256]; mx = fmaxf(mx, v[i]); }
  red[threadIdx.x] = mx; __syncthreads();
  for (int off = 128; off > 0; off >>= 1) {
    if (threadIdx.x < off) red[threadIdx.x] = fmaxf(red[threadIdx.x], red[threadIdx.x + off]);
    __syncthreads();
  }
  mx = red[0]; __syncthreads();
  float sum = 0.f;
  #pragma unroll
  for (int i = 0; i < 8; ++i) { v[i] = expf(v[i] - mx); sum += v[i]; }
  red[threadIdx.x] = sum; __syncthreads();
  for (int off = 128; off > 0; off >>= 1) {
    if (threadIdx.x < off) red[threadIdx.x] += red[threadIdx.x + off];
    __syncthreads();
  }
  float inv = 1.f / red[0];
  float* p = probs + (size_t)b * S_LEN;
  #pragma unroll
  for (int i = 0; i < 8; ++i) p[threadIdx.x + i * 256] = v[i] * inv;
}

// ---------------- att[b,d] = sum_t probs[b,t] * V[b,t,d] ----------------
__global__ __launch_bounds__(256) void attn_pv(
    const float* __restrict__ probs, const _Float16* __restrict__ vh,
    float* __restrict__ att)
{
  __shared__ float sp[256];
  int b = blockIdx.y;
  int d = blockIdx.x * 256 + threadIdx.x;
  float acc = 0.f;
  for (int t0 = 0; t0 < S_LEN; t0 += 256) {
    sp[threadIdx.x] = probs[(size_t)b * S_LEN + t0 + threadIdx.x];
    __syncthreads();
    for (int tt = 0; tt < 256; ++tt)
      acc += sp[tt] * (float)vh[((size_t)b * S_LEN + t0 + tt) * D_DIM + d];
    __syncthreads();
  }
  att[b * D_DIM + d] = acc;
}

// ---------------- LayerNorm(a + resid) * g + b, 8 rows ----------------
__global__ __launch_bounds__(256) void ln_res(
    const float* __restrict__ a, const float* __restrict__ resid, long residStride,
    const float* __restrict__ g, const float* __restrict__ be, float* __restrict__ out)
{
  __shared__ float r1[256], r2[256];
  int b = blockIdx.x;
  float v[4]; float s1 = 0.f, s2 = 0.f;
  #pragma unroll
  for (int i = 0; i < 4; ++i) {
    int d = threadIdx.x + i * 256;
    float x = a[b * D_DIM + d] + resid[(long)b * residStride + d];
    v[i] = x; s1 += x; s2 += x * x;
  }
  r1[threadIdx.x] = s1; r2[threadIdx.x] = s2; __syncthreads();
  for (int off = 128; off > 0; off >>= 1) {
    if (threadIdx.x < off) { r1[threadIdx.x] += r1[threadIdx.x + off];
                             r2[threadIdx.x] += r2[threadIdx.x + off]; }
    __syncthreads();
  }
  float mean = r1[0] * (1.f / D_DIM);
  float var  = r2[0] * (1.f / D_DIM) - mean * mean;
  float rstd = rsqrtf(var + 1e-5f);
  #pragma unroll
  for (int i = 0; i < 4; ++i) {
    int d = threadIdx.x + i * 256;
    out[b * D_DIM + d] = (v[i] - mean) * rstd * g[d] + be[d];
  }
}

// ---------------- logits[b,c] = dot(h2[b,:], Wc[:,c]) + bc[c] ----------------
__global__ __launch_bounds__(256) void classifier(
    const float* __restrict__ h2, const float* __restrict__ Wc,
    const float* __restrict__ bc, float* __restrict__ logits)
{
  __shared__ float red[256];
  int b = blockIdx.x >> 1;
  int c = blockIdx.x & 1;
  float acc = 0.f;
  for (int k = threadIdx.x; k < D_DIM; k += 256)
    acc += h2[b * D_DIM + k] * Wc[k * NCLS + c];
  red[threadIdx.x] = acc; __syncthreads();
  for (int off = 128; off > 0; off >>= 1) {
    if (threadIdx.x < off) red[threadIdx.x] += red[threadIdx.x + off];
    __syncthreads();
  }
  if (threadIdx.x == 0) logits[b * NCLS + c] = red[0] + bc[c];
}

extern "C" void kernel_launch(void* const* d_in, const int* in_sizes, int n_in,
                              void* d_out, int out_size, void* d_ws, size_t ws_size,
                              hipStream_t stream)
{
  const float* x  = (const float*)d_in[0];
  const float* Wq = (const float*)d_in[1];
  const float* bq = (const float*)d_in[2];
  const float* Wk = (const float*)d_in[3];
  const float* bk = (const float*)d_in[4];
  const float* Wv = (const float*)d_in[5];
  const float* bv = (const float*)d_in[6];
  const float* Wd = (const float*)d_in[7];
  const float* bd = (const float*)d_in[8];
  const float* g1 = (const float*)d_in[9];
  const float* b1 = (const float*)d_in[10];
  const float* g2 = (const float*)d_in[11];
  const float* b2 = (const float*)d_in[12];
  const float* Wc = (const float*)d_in[13];
  const float* bc = (const float*)d_in[14];
  float* logits = (float*)d_out;

  char* ws = (char*)d_ws;
  size_t off = 0;
  auto alloc = [&](size_t bytes) -> char* {
    char* p = ws + off; off += (bytes + 255) & ~(size_t)255; return p;
  };
  const size_t MR = (size_t)B_SZ * S_LEN;  // 16384 rows

  _Float16* xh  = (_Float16*)alloc(MR * D_DIM * 2);
  _Float16* wkh = (_Float16*)alloc((size_t)D_DIM * D_DIM * 2);
  _Float16* wvh = (_Float16*)alloc((size_t)D_DIM * D_DIM * 2);
  _Float16* kh  = (_Float16*)alloc(MR * D_DIM * 2);
  _Float16* vh  = (_Float16*)alloc(MR * D_DIM * 2);
  float* q0     = (float*)alloc((size_t)B_SZ * D_DIM * 4);
  float* scores = (float*)alloc((size_t)B_SZ * S_LEN * 4);
  float* probs  = (float*)alloc((size_t)B_SZ * S_LEN * 4);
  float* att    = (float*)alloc((size_t)B_SZ * D_DIM * 4);
  float* w1     = (float*)alloc((size_t)B_SZ * D_DIM * 4);
  float* hbuf   = (float*)alloc((size_t)B_SZ * D_DIM * 4);
  float* h2     = (float*)alloc((size_t)B_SZ * D_DIM * 4);

  // fp32 -> f16 staging for the WMMA GEMMs
  {
    int n4 = (int)(MR * D_DIM / 4);
    cvt_f32_f16<<<(n4 + 255) / 256, 256, 0, stream>>>(x, xh, n4);
    int w4 = D_DIM * D_DIM / 4;
    cvt_f32_f16<<<(w4 + 255) / 256, 256, 0, stream>>>(Wk, wkh, w4);
    cvt_f32_f16<<<(w4 + 255) / 256, 256, 0, stream>>>(Wv, wvh, w4);
  }

  // Dominant FLOPs: K = x@Wk + bk, V = x@Wv + bv (WMMA f16 -> fp32 acc)
  dim3 gg(D_DIM / BN, (unsigned)(MR / BM));
  gemm_f16_wmma<<<gg, 256, 0, stream>>>(xh, wkh, bk, kh, (int)MR, D_DIM, D_DIM);
  gemm_f16_wmma<<<gg, 256, 0, stream>>>(xh, wvh, bv, vh, (int)MR, D_DIM, D_DIM);

  // Only query row s=0 feeds the logits: tiny exact projections from here on
  proj_row<<<dim3(D_DIM / 256, B_SZ), 256, 0, stream>>>(x, (long)S_LEN * D_DIM, Wq, bq, q0, D_DIM, D_DIM, 0);
  attn_scores<<<(B_SZ * S_LEN) / 8, 256, 0, stream>>>(q0, kh, scores);
  attn_softmax<<<B_SZ, 256, 0, stream>>>(scores, probs);
  attn_pv<<<dim3(D_DIM / 256, B_SZ), 256, 0, stream>>>(probs, vh, att);
  ln_res<<<B_SZ, 256, 0, stream>>>(att, x, (long)S_LEN * D_DIM, g1, b1, w1);
  proj_row<<<dim3(D_DIM / 256, B_SZ), 256, 0, stream>>>(w1, (long)D_DIM, Wd, bd, hbuf, D_DIM, D_DIM, 1);
  ln_res<<<B_SZ, 256, 0, stream>>>(hbuf, w1, (long)D_DIM, g2, b2, h2);
  classifier<<<B_SZ * NCLS, 256, 0, stream>>>(h2, Wc, bc, logits);
}
